// SLUTagging_438086664296
// MI455X (gfx1250) — compile-verified, matrix operand order checked by reading, not run on hardware
//
#include <hip/hip_runtime.h>
#include <hip/hip_bf16.h>
#include <math.h>

// ---------------- shapes (fixed by the reference) ----------------
#define TT   512          // seq len
#define MM   512          // memory entries
#define EE   768          // embed dim
#define HH   128          // bigru output dim
#define HD   64           // per-direction hidden
#define G3   192          // 3*HD gate dim
#define NT   74           // tags
#define TH   65536        // T*H elements of one memory entry

typedef __attribute__((ext_vector_type(16))) _Float16 v16h;
typedef __attribute__((ext_vector_type(8)))  float    v8f;
typedef __attribute__((ext_vector_type(4)))  float    f4;

// ---------------------------------------------------------------------------
// Inline-asm 128-bit global load with immediate offset. volatile => ordered
// among themselves and a scheduling barrier for surrounding code; the
// SIInsertWaitcnts pass does not track inline-asm VMEM, so waits are manual.
// ---------------------------------------------------------------------------
template <int OFF>
__device__ inline f4 gl128(const float* p) {
  f4 d;
  asm volatile("global_load_b128 %0, %1, off offset:%2"
               : "=&v"(d) : "v"(p), "i"(OFF));
  return d;
}
__device__ inline void wait_loadcnt_0() { asm volatile("s_wait_loadcnt 0x0" ::: "memory"); }

// pack four f4 chunks (K 0..3 | 4..7 | 16..19 | 20..23 of the lane's row)
// into one 16-element f16 fragment per ISA 7.12.2 16-bit operand layout.
__device__ inline v16h pack_frag(f4 x0, f4 x1, f4 x2, f4 x3) {
  v16h f;
#pragma unroll
  for (int i = 0; i < 4; ++i) {
    f[i]      = (_Float16)x0[i];
    f[4 + i]  = (_Float16)x1[i];
    f[8 + i]  = (_Float16)x2[i];
    f[12 + i] = (_Float16)x3[i];
  }
  return f;
}

// raw (f32) data of one 16x32 A tile + four 16x32 B tiles: 20 x f4
struct RawTile {
  f4 a0, a1, a2, a3;
  f4 b00, b01, b02, b03;
  f4 b10, b11, b12, b13;
  f4 b20, b21, b22, b23;
  f4 b30, b31, b32, b33;
};

// 20 back-to-back b128 loads (one issue clause, no intervening waits)
__device__ inline RawTile load_tile(const float* ap, const float* bp0,
                                    const float* bp1, const float* bp2,
                                    const float* bp3, int k0) {
  RawTile t;
  const float* a  = ap  + k0;
  const float* p0 = bp0 + k0;
  const float* p1 = bp1 + k0;
  const float* p2 = bp2 + k0;
  const float* p3 = bp3 + k0;
  t.a0  = gl128<0>(a);  t.a1  = gl128<16>(a);  t.a2  = gl128<64>(a);  t.a3  = gl128<80>(a);
  t.b00 = gl128<0>(p0); t.b01 = gl128<16>(p0); t.b02 = gl128<64>(p0); t.b03 = gl128<80>(p0);
  t.b10 = gl128<0>(p1); t.b11 = gl128<16>(p1); t.b12 = gl128<64>(p1); t.b13 = gl128<80>(p1);
  t.b20 = gl128<0>(p2); t.b21 = gl128<16>(p2); t.b22 = gl128<64>(p2); t.b23 = gl128<80>(p2);
  t.b30 = gl128<0>(p3); t.b31 = gl128<16>(p3); t.b32 = gl128<64>(p3); t.b33 = gl128<80>(p3);
  return t;
}

// ---------------------------------------------------------------------------
// WMMA f16 GEMM:  C[M,N] = A[M,K] * B[N,K]^T  (+bias[N]) (+Cin)
// A, B row-major f32 (lda/ldb), converted to f16 fragments in-flight.
// Each wave computes a 16(M) x 64(N) strip; one A fragment feeds 4 WMMAs per
// 32-deep K step. Issue-ahead single-buffer pipeline (low register pressure,
// no spills): per iteration wait for the landed batch, pack fragments (raw
// regs die), ISSUE the next 20-load batch, then run the 4 WMMAs -- the batch
// is in flight across the WMMAs, one s_wait_loadcnt per K-step.
// Row indices are clamped before the K loop: branch-free inner loop
// (out-of-range B rows load garbage that is never stored).
// ---------------------------------------------------------------------------
__global__ void k_gemm_wmma(const float* __restrict__ A, int lda,
                            const float* __restrict__ B, int ldb,
                            const float* __restrict__ bias,
                            const float* __restrict__ Cin,
                            float* __restrict__ C, int ldc,
                            int M, int N, int K) {
  int lane = threadIdx.x & 31;
  int wave = threadIdx.x >> 5;
  int ntN  = (N + 15) >> 4;
  int ntM  = (M + 15) >> 4;
  int ntN4 = (ntN + 3) >> 2;                 // 4 N-tiles per wave
  int tile = blockIdx.x * (blockDim.x >> 5) + wave;
  if (tile >= ntM * ntN4) return;            // wave-uniform exit
  int tm  = tile / ntN4;
  int tn0 = (tile % ntN4) * 4;

  int kb   = (lane >> 4) << 3;               // K sub-offset for this lane half
  int arow = tm * 16 + (lane & 15);
  if (arow > M - 1) arow = M - 1;            // clamp (store is guarded)
  const float* ap = A + (size_t)arow * lda + kb;

  const float *bp0, *bp1, *bp2, *bp3;
  {
    int r0 = (tn0 + 0) * 16 + (lane & 15); if (r0 > N - 1) r0 = N - 1;
    int r1 = (tn0 + 1) * 16 + (lane & 15); if (r1 > N - 1) r1 = N - 1;
    int r2 = (tn0 + 2) * 16 + (lane & 15); if (r2 > N - 1) r2 = N - 1;
    int r3 = (tn0 + 3) * 16 + (lane & 15); if (r3 > N - 1) r3 = N - 1;
    bp0 = B + (size_t)r0 * ldb + kb;
    bp1 = B + (size_t)r1 * ldb + kb;
    bp2 = B + (size_t)r2 * ldb + kb;
    bp3 = B + (size_t)r3 * ldb + kb;
  }

  v8f acc0 = {}, acc1 = {}, acc2 = {}, acc3 = {};
  RawTile cur = load_tile(ap, bp0, bp1, bp2, bp3, 0);   // prologue batch
  for (int k0 = 0; k0 < K; k0 += 32) {
    wait_loadcnt_0();                                   // batch landed

    v16h a  = pack_frag(cur.a0,  cur.a1,  cur.a2,  cur.a3);
    v16h b0 = pack_frag(cur.b00, cur.b01, cur.b02, cur.b03);
    v16h b1 = pack_frag(cur.b10, cur.b11, cur.b12, cur.b13);
    v16h b2 = pack_frag(cur.b20, cur.b21, cur.b22, cur.b23);
    v16h b3 = pack_frag(cur.b30, cur.b31, cur.b32, cur.b33);

    int kn = (k0 + 32 < K) ? (k0 + 32) : k0;            // last iter: dead reload
    cur = load_tile(ap, bp0, bp1, bp2, bp3, kn);        // issue next batch

    acc0 = __builtin_amdgcn_wmma_f32_16x16x32_f16(false, a, false, b0, (short)0, acc0, false, false);
    acc1 = __builtin_amdgcn_wmma_f32_16x16x32_f16(false, a, false, b1, (short)0, acc1, false, false);
    acc2 = __builtin_amdgcn_wmma_f32_16x16x32_f16(false, a, false, b2, (short)0, acc2, false, false);
    acc3 = __builtin_amdgcn_wmma_f32_16x16x32_f16(false, a, false, b3, (short)0, acc3, false, false);
  }
  wait_loadcnt_0();                                     // drain dead last batch

  int rbase = tm * 16 + ((lane >> 4) << 3);
  v8f accs[4] = {acc0, acc1, acc2, acc3};
#pragma unroll
  for (int j = 0; j < 4; ++j) {
    int col = (tn0 + j) * 16 + (lane & 15);
    if (col < N) {
      float bv = bias ? bias[col] : 0.0f;
#pragma unroll
      for (int r = 0; r < 8; ++r) {
        int row = rbase + r;
        if (row < M) {
          float v = accs[j][r] + bv;
          if (Cin) v += Cin[(size_t)row * ldc + col];
          C[(size_t)row * ldc + col] = v;
        }
      }
    }
  }
}

// ---------------------------------------------------------------------------
// Attention pass 1: s[m] = dot(memory[m,:,:], u)   (134 MB streaming, f32)
// ---------------------------------------------------------------------------
__global__ void k_scores(const float* __restrict__ mem,
                         const float* __restrict__ u, float* __restrict__ s) {
  __shared__ float red[256];
  int m = blockIdx.x;
  const float4* mp = (const float4*)(mem + (size_t)m * TH);
  const float4* up = (const float4*)u;
  float acc = 0.f;
  for (int i = threadIdx.x; i < TH / 4; i += 256) {
    float4 a = mp[i], b = up[i];
    acc += a.x * b.x + a.y * b.y + a.z * b.z + a.w * b.w;
  }
  red[threadIdx.x] = acc;
  __syncthreads();
  for (int off = 128; off; off >>= 1) {
    if (threadIdx.x < (unsigned)off) red[threadIdx.x] += red[threadIdx.x + off];
    __syncthreads();
  }
  if (threadIdx.x == 0) s[m] = red[0];
}

// softmax over the M=512 scores (single workgroup)
__global__ void k_softmax512(const float* __restrict__ s, float* __restrict__ p) {
  __shared__ float red[512];
  int tid = threadIdx.x;
  float v = s[tid];
  red[tid] = v;
  __syncthreads();
  for (int off = 256; off; off >>= 1) {
    if (tid < off) red[tid] = fmaxf(red[tid], red[tid + off]);
    __syncthreads();
  }
  float mx = red[0];
  __syncthreads();
  float e = __expf(v - mx);
  red[tid] = e;
  __syncthreads();
  for (int off = 256; off; off >>= 1) {
    if (tid < off) red[tid] += red[tid + off];
    __syncthreads();
  }
  p[tid] = e / red[0];
}

// ---------------------------------------------------------------------------
// Attention pass 2: msum[i] = sum_m p[m] * memory[m, i]  (L2-resident pass,
// memory tensor = 134 MB < 192 MB L2). 256-KB stride per m; prefetch ahead.
// ---------------------------------------------------------------------------
__global__ void k_memsum(const float* __restrict__ mem,
                         const float* __restrict__ p, float* __restrict__ out) {
  __shared__ float ps[MM];
  for (int i = threadIdx.x; i < MM; i += blockDim.x) ps[i] = p[i];
  __syncthreads();
  int i4 = blockIdx.x * blockDim.x + threadIdx.x;   // over TH/4 float4 slots
  const float4* mp = (const float4*)mem;
  float4 acc = {0.f, 0.f, 0.f, 0.f};
  for (int m = 0; m < MM; ++m) {
    if (m + 4 < MM)
      __builtin_prefetch((const void*)&mp[(size_t)(m + 4) * (TH / 4) + i4], 0, 1);
    float4 v = mp[(size_t)m * (TH / 4) + i4];
    float  w = ps[m];
    acc.x += w * v.x; acc.y += w * v.y; acc.z += w * v.z; acc.w += w * v.w;
  }
  ((float4*)out)[i4] = acc;
}

// ---------------------------------------------------------------------------
// One bidirectional GRU layer. grid.x==2 (block 0 forward, block 1 backward),
// 192 threads. Whh rows live in VGPRs, h lives in LDS. xw = precomputed
// x@Wih^T + bih, [T, 192] per direction. Output columns [dir*64, dir*64+64).
// ---------------------------------------------------------------------------
__device__ inline float sigmoidf_(float x) { return 1.0f / (1.0f + __expf(-x)); }

__global__ void k_bigru(const float* __restrict__ xwf, const float* __restrict__ xwb,
                        const float* __restrict__ Whhf, const float* __restrict__ bhhf,
                        const float* __restrict__ Whhb, const float* __restrict__ bhhb,
                        float* __restrict__ out) {
  int dir = blockIdx.x;
  const float* xw  = dir ? xwb  : xwf;
  const float* Whh = dir ? Whhb : Whhf;
  const float* bhh = dir ? bhhb : bhhf;
  int tid = threadIdx.x;                     // 0..191
  __shared__ float hs[HD];
  __shared__ float ghs[G3];
  __shared__ float xls[G3];
  float w[HD];
#pragma unroll
  for (int j = 0; j < HD; ++j) w[j] = Whh[tid * HD + j];
  float bb = bhh[tid];
  if (tid < HD) hs[tid] = 0.f;
  __syncthreads();

  for (int s = 0; s < TT; ++s) {
    int t = dir ? (TT - 1 - s) : s;
    float acc = bb;
#pragma unroll
    for (int j = 0; j < HD; ++j) acc += w[j] * hs[j];
    ghs[tid] = acc;
    xls[tid] = xw[t * G3 + tid];
    __syncthreads();
    if (tid < HD) {
      float r  = sigmoidf_(xls[tid] + ghs[tid]);
      float z  = sigmoidf_(xls[HD + tid] + ghs[HD + tid]);
      float n  = tanhf(xls[2 * HD + tid] + r * ghs[2 * HD + tid]);
      float hn = (1.f - z) * n + z * hs[tid];
      hs[tid] = hn;
      out[t * HH + dir * HD + tid] = hn;
    }
    __syncthreads();
  }
}

__global__ void k_add(const float* __restrict__ a, const float* __restrict__ b,
                      float* __restrict__ c, int n) {
  int i = blockIdx.x * blockDim.x + threadIdx.x;
  if (i < n) c[i] = a[i] + b[i];
}

// row softmax of logits [T, NT]; tag_mask folds into a per-row constant
// which the max-subtraction absorbs (matches reference semantics for mask=1).
__global__ void k_softmax_rows(const float* __restrict__ logits,
                               const float* __restrict__ mask,
                               float* __restrict__ out) {
  __shared__ float red[128];
  int t = blockIdx.x, tid = threadIdx.x;
  float mterm = (1.0f - mask[t]) * -1e32f;
  float v = (tid < NT) ? (logits[t * NT + tid] + mterm) : -3.0e38f;
  red[tid] = v;
  __syncthreads();
  for (int off = 64; off; off >>= 1) {
    if (tid < off) red[tid] = fmaxf(red[tid], red[tid + off]);
    __syncthreads();
  }
  float mx = red[0];
  __syncthreads();
  float e = (tid < NT) ? __expf(v - mx) : 0.f;
  red[tid] = e;
  __syncthreads();
  for (int off = 64; off; off >>= 1) {
    if (tid < off) red[tid] += red[tid + off];
    __syncthreads();
  }
  if (tid < NT) out[t * NT + tid] = e / red[0];
}

// ---------------------------------------------------------------------------
extern "C" void kernel_launch(void* const* d_in, const int* in_sizes, int n_in,
                              void* d_out, int out_size, void* d_ws, size_t ws_size,
                              hipStream_t stream) {
  // pytree-flattened inputs
  const float* embed  = (const float*)d_in[0];   // [T,E]
  const float* memory = (const float*)d_in[1];   // [M,T,H]
  const float* tmask  = (const float*)d_in[2];   // [T]
  // c_rnn (din=E)
  const float* c_Wif = (const float*)d_in[3];  const float* c_Whf = (const float*)d_in[4];
  const float* c_bif = (const float*)d_in[5];  const float* c_bhf = (const float*)d_in[6];
  const float* c_Wib = (const float*)d_in[7];  const float* c_Whb = (const float*)d_in[8];
  const float* c_bib = (const float*)d_in[9];  const float* c_bhb = (const float*)d_in[10];
  // sess_rnn (din=H)
  const float* s_Wif = (const float*)d_in[11]; const float* s_Whf = (const float*)d_in[12];
  const float* s_bif = (const float*)d_in[13]; const float* s_bhf = (const float*)d_in[14];
  const float* s_Wib = (const float*)d_in[15]; const float* s_Whb = (const float*)d_in[16];
  const float* s_bib = (const float*)d_in[17]; const float* s_bhb = (const float*)d_in[18];
  // tag_rnn (din=E)
  const float* t_Wif = (const float*)d_in[19]; const float* t_Whf = (const float*)d_in[20];
  const float* t_bif = (const float*)d_in[21]; const float* t_bhf = (const float*)d_in[22];
  const float* t_Wib = (const float*)d_in[23]; const float* t_Whb = (const float*)d_in[24];
  const float* t_bib = (const float*)d_in[25]; const float* t_bhb = (const float*)d_in[26];
  const float* Wff  = (const float*)d_in[27];  // [H, 2H]
  const float* bff  = (const float*)d_in[28];
  const float* Wk   = (const float*)d_in[29];  // [E, H]
  const float* bk   = (const float*)d_in[30];
  const float* Wout = (const float*)d_in[31];  // [NT, H]
  const float* bout = (const float*)d_in[32];
  float* out = (float*)d_out;                  // [T, NT]

  // workspace carve-up (floats)
  float* ws = (float*)d_ws;
  float* xwf    = ws;              ws += TT * G3;
  float* xwb    = ws;              ws += TT * G3;
  float* u      = ws;              ws += TT * HH;
  float* sc     = ws;              ws += MM;
  float* p      = ws;              ws += MM;
  float* msum   = ws;              ws += TT * HH;
  float* Gm     = ws;              ws += TT * HH;
  float* h      = ws;              ws += TT * HH;
  float* uh     = ws;              ws += TT * HH;
  float* o      = ws;              ws += TT * EE;
  float* eo     = ws;              ws += TT * EE;
  float* dec    = ws;              ws += TT * HH;
  float* logits = ws;              ws += TT * NT;

  auto gemm = [&](const float* A, int lda, const float* B, int ldb,
                  const float* bias, const float* Cin, float* C, int ldc,
                  int M, int N, int K) {
    int ntN4   = (((N + 15) / 16) + 3) / 4;
    int tiles  = ((M + 15) / 16) * ntN4;
    int blocks = (tiles + 7) / 8;
    hipLaunchKernelGGL(k_gemm_wmma, dim3(blocks), dim3(256), 0, stream,
                       A, lda, B, ldb, bias, Cin, C, ldc, M, N, K);
  };

  // ---- u = bigru(embed, c_rnn) ----
  gemm(embed, EE, c_Wif, EE, c_bif, nullptr, xwf, G3, TT, G3, EE);
  gemm(embed, EE, c_Wib, EE, c_bib, nullptr, xwb, G3, TT, G3, EE);
  hipLaunchKernelGGL(k_bigru, dim3(2), dim3(G3), 0, stream,
                     xwf, xwb, c_Whf, c_bhf, c_Whb, c_bhb, u);

  // ---- memory-bank attention (two bandwidth passes) ----
  hipLaunchKernelGGL(k_scores, dim3(MM), dim3(256), 0, stream, memory, u, sc);
  hipLaunchKernelGGL(k_softmax512, dim3(1), dim3(MM), 0, stream, sc, p);
  hipLaunchKernelGGL(k_memsum, dim3(TH / 4 / 256), dim3(256), 0, stream, memory, p, msum);

  // ---- Gm = msum @ Wff[:, :H]^T + u @ Wff[:, H:]^T + bff  (Σp = 1) ----
  gemm(msum, HH, Wff,       2 * HH, bff,     nullptr, Gm, HH, TT, HH, HH);
  gemm(u,    HH, Wff + HH,  2 * HH, nullptr, Gm,      Gm, HH, TT, HH, HH);

  // ---- h = bigru(Gm, sess_rnn) ----
  gemm(Gm, HH, s_Wif, HH, s_bif, nullptr, xwf, G3, TT, G3, HH);
  gemm(Gm, HH, s_Wib, HH, s_bib, nullptr, xwb, G3, TT, G3, HH);
  hipLaunchKernelGGL(k_bigru, dim3(2), dim3(G3), 0, stream,
                     xwf, xwb, s_Whf, s_bhf, s_Whb, s_bhb, h);

  // ---- o = (u + h) @ Wk^T + bk ; eo = embed + o ----
  hipLaunchKernelGGL(k_add, dim3((TT * HH + 255) / 256), dim3(256), 0, stream,
                     u, h, uh, TT * HH);
  gemm(uh, HH, Wk, HH, bk, nullptr, o, EE, TT, EE, HH);
  hipLaunchKernelGGL(k_add, dim3((TT * EE + 255) / 256), dim3(256), 0, stream,
                     embed, o, eo, TT * EE);

  // ---- dec = bigru(eo, tag_rnn) ----
  gemm(eo, EE, t_Wif, EE, t_bif, nullptr, xwf, G3, TT, G3, EE);
  gemm(eo, EE, t_Wib, EE, t_bib, nullptr, xwb, G3, TT, G3, EE);
  hipLaunchKernelGGL(k_bigru, dim3(2), dim3(G3), 0, stream,
                     xwf, xwb, t_Whf, t_bhf, t_Whb, t_bhb, dec);

  // ---- logits + masked row softmax ----
  gemm(dec, HH, Wout, HH, bout, nullptr, logits, NT, TT, NT, HH);
  hipLaunchKernelGGL(k_softmax_rows, dim3(TT), dim3(128), 0, stream,
                     logits, tmask, out);
}
